// GroupedQueryAttention_75204877353288
// MI455X (gfx1250) — compile-verified, matrix operand order checked by reading, not run on hardware
//
#include <hip/hip_runtime.h>

// ---------------- problem constants ----------------
#define B_    2
#define S_    2048
#define D_    2048
#define H_    16
#define KV_   4
#define G_    4          // NUM_GROUPS = H_/KV_
#define HD_   128        // head dim
#define MROWS (B_*S_)    // 4096 token rows
#define KVD   (KV_*HD_)  // 512

typedef __bf16 bf16;
typedef __attribute__((ext_vector_type(16))) bf16  v16bf;
typedef __attribute__((ext_vector_type(8)))  bf16  v8bf;
typedef __attribute__((ext_vector_type(8)))  float v8f;

// Load a 16x32 bf16 A/B fragment slice for this lane.
// Layout (ISA 7.12.2): lanes 0-15 hold K 0..7 (lo) and 16..23 (hi);
// lanes 16-31 hold K 8..15 and 24..31. Caller pre-adds (lane>=16 ? 8 : 0).
static __device__ inline v16bf load_frag16(const bf16* p) {
  union { v16bf v; v8bf h[2]; } u;
  u.h[0] = *(const v8bf*)(p);
  u.h[1] = *(const v8bf*)(p + 16);
  return u.v;
}

static __device__ inline v8f vzero8() {
  v8f z;
  for (int i = 0; i < 8; ++i) z[i] = 0.0f;
  return z;
}

static __device__ inline v8f wmma_bf16(v16bf a, v16bf b, v8f c) {
  // (neg_a, A, neg_b, B, c_mod, C, reuse_a, reuse_b)
  return __builtin_amdgcn_wmma_f32_16x16x32_bf16(false, a, false, b,
                                                 (short)0, c, false, false);
}

// ---------------- conversion kernels ----------------
__global__ void cvt_bf16(const float* __restrict__ in, bf16* __restrict__ out, int n) {
  int i = blockIdx.x * blockDim.x + threadIdx.x;
  if (i < n) out[i] = (bf16)in[i];
}

// in: R x Cn fp32 row-major  ->  out: Cn x R bf16 row-major (transposed)
__global__ void cvt_transpose_bf16(const float* __restrict__ in, bf16* __restrict__ out,
                                   int R, int Cn) {
  __shared__ float tile[16][17];
  int tx = threadIdx.x, ty = threadIdx.y;
  int c0 = blockIdx.x * 16, r0 = blockIdx.y * 16;
  tile[ty][tx] = in[(size_t)(r0 + ty) * Cn + (c0 + tx)];
  __syncthreads();
  out[(size_t)(c0 + ty) * R + (r0 + tx)] = (bf16)tile[tx][ty];
}

// ---------------- WMMA GEMM: C[M,N] = A[M,K] * Bt[N,K]^T ----------------
// 256 threads = 8 waves in a 2x4 grid; each wave computes a 32x64 C tile
// (2 A-frags x 4 B-frags -> 8 WMMAs per K=32 step, 6 fragment loads).
// K-loop unrolled by 2 with ping-pong fragment buffers: no register copies,
// loads for one step issue while the other step's 8 WMMAs execute.
// Requires (K/32) >= 2; all calls here use K = 2048 (64 even steps).
__global__ __launch_bounds__(256) void gemm_bf16_nt(const bf16* __restrict__ A,
                                                    const bf16* __restrict__ Bt,
                                                    float* __restrict__ C,
                                                    int M, int N, int K) {
  const int lane = threadIdx.x & 31;
  const int w    = threadIdx.x >> 5;
  const int hl   = lane >> 4;
  const int l16  = lane & 15;
  const int sel8 = hl * 8;
  const int wm   = w & 1;       // 0..1
  const int wn   = w >> 1;      // 0..3
  const int rowBase = blockIdx.y * 64 + wm * 32;
  const int colBase = blockIdx.x * 256 + wn * 64;

  v8f acc[2][4];
#pragma unroll
  for (int im = 0; im < 2; ++im)
#pragma unroll
    for (int j = 0; j < 4; ++j) acc[im][j] = vzero8();

  const bf16* arow0 = A + (size_t)(rowBase + l16) * K + sel8;
  const bf16* arow1 = arow0 + (size_t)16 * K;
  const bf16* brow[4];
#pragma unroll
  for (int j = 0; j < 4; ++j)
    brow[j] = Bt + (size_t)(colBase + j * 16 + l16) * K + sel8;

  // ping-pong fragment buffers
  v16bf a0A, a1A, bbA[4];
  v16bf a0B, a1B, bbB[4];

  // prologue: step 0 -> buffer A
  a0A = load_frag16(arow0);
  a1A = load_frag16(arow1);
#pragma unroll
  for (int j = 0; j < 4; ++j) bbA[j] = load_frag16(brow[j]);

  int k = 32;
  for (; k + 64 <= K; k += 64) {
    // loads for step k -> buffer B
    a0B = load_frag16(arow0 + k);
    a1B = load_frag16(arow1 + k);
#pragma unroll
    for (int j = 0; j < 4; ++j) bbB[j] = load_frag16(brow[j] + k);
    // consume buffer A (step k-32)
#pragma unroll
    for (int j = 0; j < 4; ++j) acc[0][j] = wmma_bf16(a0A, bbA[j], acc[0][j]);
#pragma unroll
    for (int j = 0; j < 4; ++j) acc[1][j] = wmma_bf16(a1A, bbA[j], acc[1][j]);
    // loads for step k+32 -> buffer A
    a0A = load_frag16(arow0 + k + 32);
    a1A = load_frag16(arow1 + k + 32);
#pragma unroll
    for (int j = 0; j < 4; ++j) bbA[j] = load_frag16(brow[j] + k + 32);
    // consume buffer B (step k)
#pragma unroll
    for (int j = 0; j < 4; ++j) acc[0][j] = wmma_bf16(a0B, bbB[j], acc[0][j]);
#pragma unroll
    for (int j = 0; j < 4; ++j) acc[1][j] = wmma_bf16(a1B, bbB[j], acc[1][j]);
  }
  if (k < K) {
    // one more step (k): load into buffer B, then drain both buffers
    a0B = load_frag16(arow0 + k);
    a1B = load_frag16(arow1 + k);
#pragma unroll
    for (int j = 0; j < 4; ++j) bbB[j] = load_frag16(brow[j] + k);
#pragma unroll
    for (int j = 0; j < 4; ++j) acc[0][j] = wmma_bf16(a0A, bbA[j], acc[0][j]);
#pragma unroll
    for (int j = 0; j < 4; ++j) acc[1][j] = wmma_bf16(a1A, bbA[j], acc[1][j]);
#pragma unroll
    for (int j = 0; j < 4; ++j) acc[0][j] = wmma_bf16(a0B, bbB[j], acc[0][j]);
#pragma unroll
    for (int j = 0; j < 4; ++j) acc[1][j] = wmma_bf16(a1B, bbB[j], acc[1][j]);
  } else {
    // drain buffer A
#pragma unroll
    for (int j = 0; j < 4; ++j) acc[0][j] = wmma_bf16(a0A, bbA[j], acc[0][j]);
#pragma unroll
    for (int j = 0; j < 4; ++j) acc[1][j] = wmma_bf16(a1A, bbA[j], acc[1][j]);
  }

  // C layout: VGPR i -> row i (+8 for lanes 16..31), lane&15 -> column
#pragma unroll
  for (int im = 0; im < 2; ++im)
#pragma unroll
    for (int j = 0; j < 4; ++j)
#pragma unroll
      for (int i = 0; i < 8; ++i)
        C[(size_t)(rowBase + im * 16 + i + hl * 8) * N +
          (colBase + j * 16 + l16)] = acc[im][j][i];
}

// ---------------- RoPE + head-major repack (fp32 -> bf16) ----------------
// in:  Qraw[(b*S+s), head*128+d]   out: Qh[((b*nH+head)*S+s)*128+d] with RoPE
__global__ void rope_to_heads(const float* __restrict__ Qraw, bf16* __restrict__ Qh, int nH) {
  int idx = blockIdx.x * blockDim.x + threadIdx.x;
  int total = B_ * S_ * nH * HD_;
  if (idx >= total) return;
  int d  = idx & (HD_ - 1);
  int t  = idx >> 7;
  int hh = t % nH;
  int t2 = t / nH;
  int s  = t2 % S_;
  int b  = t2 / S_;
  const float* row = Qraw + (size_t)(b * S_ + s) * (nH * HD_) + hh * HD_;
  float q  = row[d];
  float q2 = row[d ^ 64];
  float fr = __powf(10000.0f, -(float)(d & 63) * (1.0f / 64.0f));
  float sn, cs;
  __sincosf((float)s * fr, &sn, &cs);
  float val = q * cs + ((d < 64) ? -q2 : q2) * sn;
  Qh[((size_t)(b * nH + hh) * S_ + s) * HD_ + d] = (bf16)val;
}

// V: (b*S+s, kv*128+d) fp32 -> Vt[((b*KV+kv)*128+d)*S + s] bf16 (transposed per head)
__global__ void v_to_t(const float* __restrict__ Vraw, bf16* __restrict__ Vt) {
  int idx = blockIdx.x * blockDim.x + threadIdx.x;
  int total = B_ * S_ * KV_ * HD_;
  if (idx >= total) return;
  int d  = idx & (HD_ - 1);
  int t  = idx >> 7;
  int kv = t % KV_;
  int t2 = t / KV_;
  int s  = t2 % S_;
  int b  = t2 / S_;
  float v = Vraw[(size_t)(b * S_ + s) * KVD + kv * HD_ + d];
  Vt[((size_t)(b * KV_ + kv) * HD_ + d) * S_ + s] = (bf16)v;
}

// ---------------- flash attention ----------------
// grid: (S/128, B*H), 256 threads = 8 waves. Each wave: 16 q-rows x head_dim 128.
__global__ __launch_bounds__(256) void flash_attn(const bf16* __restrict__ Qb,
                                                  const bf16* __restrict__ Kb,
                                                  const bf16* __restrict__ Vt,
                                                  const int* __restrict__ amask,
                                                  bf16* __restrict__ Ob) {
  __shared__ __attribute__((aligned(32))) bf16 plds[8 * 16 * 32];
  const int lane = threadIdx.x & 31;
  const int w    = threadIdx.x >> 5;
  const int hl   = lane >> 4;
  const int l16  = lane & 15;
  const int sel8 = hl * 8;

  const int bh  = blockIdx.y;   // b*H + h
  const int b   = bh / H_;
  const int h   = bh % H_;
  const int kvh = h / G_;
  const int qRow0 = blockIdx.x * 128 + w * 16;

  const bf16* Qh = Qb + (size_t)bh * S_ * HD_;
  const bf16* Kh = Kb + (size_t)(b * KV_ + kvh) * S_ * HD_;
  const bf16* Vh = Vt + (size_t)(b * KV_ + kvh) * HD_ * S_;
  const int* mrow = amask + b * S_;
  bf16* pl = plds + w * (16 * 32);

  // resident Q fragments: 16 rows x 128 head-dim = 4 K-chunks of 32
  v16bf qf[4];
#pragma unroll
  for (int kc = 0; kc < 4; ++kc)
    qf[kc] = load_frag16(Qh + (size_t)(qRow0 + l16) * HD_ + kc * 32 + sel8);

  v8f acc[8];
  float m[8], l[8];
#pragma unroll
  for (int t = 0; t < 8; ++t) acc[t] = vzero8();
#pragma unroll
  for (int i = 0; i < 8; ++i) { m[i] = -3.0e38f; l[i] = 0.0f; }

  const float scale = 0.08838834764831845f;  // 1/sqrt(128)

  for (int j0 = 0; j0 < S_; j0 += 32) {
    // load all 8 K fragments up front -> one clause, 8 back-to-back WMMAs
    v16bf kf[8];
#pragma unroll
    for (int nt = 0; nt < 2; ++nt)
#pragma unroll
      for (int kc = 0; kc < 4; ++kc)
        kf[nt * 4 + kc] =
            load_frag16(Kh + (size_t)(j0 + nt * 16 + l16) * HD_ + kc * 32 + sel8);

    v8f sc0 = vzero8(), sc1 = vzero8();
#pragma unroll
    for (int kc = 0; kc < 4; ++kc) sc0 = wmma_bf16(qf[kc], kf[kc], sc0);
#pragma unroll
    for (int kc = 0; kc < 4; ++kc) sc1 = wmma_bf16(qf[kc], kf[4 + kc], sc1);

    // issue V loads now so they are in flight during the softmax VALU work
    v16bf vf[8];
#pragma unroll
    for (int t = 0; t < 8; ++t)
      vf[t] = load_frag16(Vh + (size_t)(t * 16 + l16) * S_ + j0 + sel8);

    int mk0 = mrow[j0 + l16];
    int mk1 = mrow[j0 + 16 + l16];
#pragma unroll
    for (int i = 0; i < 8; ++i) {
      sc0[i] = mk0 ? sc0[i] * scale : -3.0e38f;
      sc1[i] = mk1 ? sc1[i] * scale : -3.0e38f;
    }
    // row-wise max over the 16 lanes of this half (rows live per-VGPR-element)
    float mx[8];
#pragma unroll
    for (int i = 0; i < 8; ++i) {
      float v = fmaxf(sc0[i], sc1[i]);
#pragma unroll
      for (int off = 1; off < 16; off <<= 1)
        v = fmaxf(v, __shfl_xor(v, off, 32));
      mx[i] = v;
    }
#pragma unroll
    for (int i = 0; i < 8; ++i) {
      float mn   = fmaxf(m[i], mx[i]);
      float corr = __expf(m[i] - mn);
      m[i] = mn;
      l[i] *= corr;
#pragma unroll
      for (int t = 0; t < 8; ++t) acc[t][i] *= corr;
    }
    // probabilities + row sums; spill P (C-layout) to per-wave LDS tile
#pragma unroll
    for (int i = 0; i < 8; ++i) {
      float p0 = __expf(sc0[i] - m[i]);
      float p1 = __expf(sc1[i] - m[i]);
      float rs = p0 + p1;
#pragma unroll
      for (int off = 1; off < 16; off <<= 1)
        rs += __shfl_xor(rs, off, 32);
      l[i] += rs;
      pl[(i + hl * 8) * 32 + l16]      = (bf16)p0;
      pl[(i + hl * 8) * 32 + 16 + l16] = (bf16)p1;
    }
    asm volatile("s_wait_dscnt 0" ::: "memory");   // wave's LDS stores visible
    // reload P in A-fragment layout (WAR on pl is safe: LDS is in-order per wave)
    v16bf pf = load_frag16(pl + l16 * 32 + sel8);
    // O += P(16x32) @ V(32x128): 8 N-tiles, B-frag rows come from transposed V
#pragma unroll
    for (int t = 0; t < 8; ++t) acc[t] = wmma_bf16(pf, vf[t], acc[t]);
  }

  float inv[8];
#pragma unroll
  for (int i = 0; i < 8; ++i) inv[i] = 1.0f / l[i];
#pragma unroll
  for (int t = 0; t < 8; ++t)
#pragma unroll
    for (int i = 0; i < 8; ++i) {
      int srow = qRow0 + i + hl * 8;
      Ob[(size_t)(b * S_ + srow) * D_ + h * HD_ + t * 16 + l16] =
          (bf16)(acc[t][i] * inv[i]);
    }
}

// ---------------- host launch ----------------
extern "C" void kernel_launch(void* const* d_in, const int* in_sizes, int n_in,
                              void* d_out, int out_size, void* d_ws, size_t ws_size,
                              hipStream_t stream) {
  (void)in_sizes; (void)n_in; (void)out_size; (void)ws_size;
  const float* x     = (const float*)d_in[0];
  const int*   amask = (const int*)d_in[1];
  const float* Wq    = (const float*)d_in[2];
  const float* Wk    = (const float*)d_in[3];
  const float* Wv    = (const float*)d_in[4];
  const float* Wo    = (const float*)d_in[5];
  float* out = (float*)d_out;

  char* ws = (char*)d_ws;
  size_t off = 0;
  auto alloc = [&](size_t bytes) -> void* {
    void* p = ws + off;
    off += (bytes + 255) & ~(size_t)255;
    return p;
  };
  bf16*  xb   = (bf16*)alloc((size_t)MROWS * D_ * 2);
  bf16*  WqT  = (bf16*)alloc((size_t)D_ * D_ * 2);
  bf16*  WkT  = (bf16*)alloc((size_t)KVD * D_ * 2);
  bf16*  WvT  = (bf16*)alloc((size_t)KVD * D_ * 2);
  bf16*  WoT  = (bf16*)alloc((size_t)D_ * D_ * 2);
  float* Qraw = (float*)alloc((size_t)MROWS * D_ * 4);
  float* Kraw = (float*)alloc((size_t)MROWS * KVD * 4);
  float* Vraw = (float*)alloc((size_t)MROWS * KVD * 4);
  bf16*  Qh   = (bf16*)alloc((size_t)MROWS * D_ * 2);
  bf16*  Kh   = (bf16*)alloc((size_t)MROWS * KVD * 2);
  bf16*  Vt   = (bf16*)alloc((size_t)MROWS * KVD * 2);
  bf16*  Ob   = (bf16*)Qraw;  // reuse: Qraw dead after RoPE repack

  // 1) precision conversion + weight transposes
  {
    int n = MROWS * D_;
    cvt_bf16<<<(n + 255) / 256, 256, 0, stream>>>(x, xb, n);
  }
  {
    dim3 t(16, 16);
    dim3 gq(D_ / 16, D_ / 16);
    cvt_transpose_bf16<<<gq, t, 0, stream>>>(Wq, WqT, D_, D_);
    dim3 gk(KVD / 16, D_ / 16);
    cvt_transpose_bf16<<<gk, t, 0, stream>>>(Wk, WkT, D_, KVD);
    cvt_transpose_bf16<<<gk, t, 0, stream>>>(Wv, WvT, D_, KVD);
    cvt_transpose_bf16<<<gq, t, 0, stream>>>(Wo, WoT, D_, D_);
  }
  // 2) QKV projections (WMMA)
  {
    dim3 g(D_ / 256, MROWS / 64);
    gemm_bf16_nt<<<g, 256, 0, stream>>>(xb, WqT, Qraw, MROWS, D_, D_);
  }
  {
    dim3 g(KVD / 256, MROWS / 64);
    gemm_bf16_nt<<<g, 256, 0, stream>>>(xb, WkT, Kraw, MROWS, KVD, D_);
    gemm_bf16_nt<<<g, 256, 0, stream>>>(xb, WvT, Vraw, MROWS, KVD, D_);
  }
  // 3) RoPE + layout repack to per-head bf16
  {
    int nq = B_ * S_ * H_ * HD_;
    rope_to_heads<<<(nq + 255) / 256, 256, 0, stream>>>(Qraw, Qh, H_);
    int nk = B_ * S_ * KV_ * HD_;
    rope_to_heads<<<(nk + 255) / 256, 256, 0, stream>>>(Kraw, Kh, KV_);
    v_to_t<<<(nk + 255) / 256, 256, 0, stream>>>(Vraw, Vt);
  }
  // 4) flash attention (WMMA)
  {
    dim3 g(S_ / 128, B_ * H_);
    flash_attn<<<g, 256, 0, stream>>>(Qh, Kh, Vt, amask, Ob);
  }
  // 5) output projection (WMMA) -> fp32 d_out
  {
    dim3 g(D_ / 256, MROWS / 64);
    gemm_bf16_nt<<<g, 256, 0, stream>>>(Ob, WoT, out, MROWS, D_, D_);
  }
}
